// QuantizedEncoder_52802327937272
// MI455X (gfx1250) — matrix operand order repaired and checked by reading, not compiled, not run on hardware
//
#include <hip/hip_runtime.h>
#include <hip/hip_fp16.h>

// ---------------------------------------------------------------------------
// Quantized encoder for MI455X (gfx1250, wave32).
//   x[256,3,64,64] -> quant_act(16) -> int8 conv 3x3 s2 p1 -> relu
//   -> quant_act(11) -> flatten [256,65536] -> int8 dense [256,65536] -> relu
//
// GEMM on v_wmma_f32_16x16x32_f16: quantized ints (<=1023 / <=127) are exact
// in fp16; f32 accumulation matches the fp32 reference einsum semantics.
// Conv output is recomputed (27 MACs, x resident in L2) instead of staging a
// 67MB f32 buffer through HBM: saves ~134MB of traffic (~5.7us @ 23.3TB/s).
// GEMM uses 16x32 register blocking (A fragment reused by 2 WMMAs) and
// split-K x8 (1024 waves) for latency hiding, + fp32 partials reduce.
// ---------------------------------------------------------------------------

typedef __attribute__((ext_vector_type(16))) _Float16 v16h;
typedef __attribute__((ext_vector_type(8)))  _Float16 v8h;
typedef __attribute__((ext_vector_type(8)))  float    v8f;

#define EPSQ 1e-8f

// Problem dimensions (fixed by the reference)
static const int BATCH = 256;
static const int CIN   = 3,  HIN = 64, WIN = 64;
static const int COUT  = 64, OH  = 32, OW  = 32;
static const int FIN   = COUT * OH * OW;   // 65536
static const int FOUT  = 256;
static const long long HN = (long long)BATCH * COUT * OH * OW; // 16,777,216

// GEMM decomposition
static const int KSPLIT   = 8;                  // K split factor
static const int KCHUNK   = FIN / KSPLIT;       // 8192
static const int NTILE_M  = BATCH / 16;         // 16  (16-row tiles)
static const int NTILE_N  = FOUT / 32;          // 8   (32-col tiles)
static const int TILES    = NTILE_M * NTILE_N;  // 128 tiles
static const int GEMM_WAVES = TILES * KSPLIT;   // 1024 waves

// Workspace layout (bytes)
static const size_t OFF_SCAL      = 0;          // [0]=absmax_x, [1]=absmax_h
static const size_t OFF_CONV_WSF  = 256;        // 64 floats
static const size_t OFF_CONV_WINT = 1024;       // 64*27 floats
static const size_t OFF_DENSE_WSF = 8192;       // 256 floats
static const size_t OFF_H_F16     = 16384;                         // HN halves
static const size_t OFF_W_F16     = OFF_H_F16 + (size_t)HN * 2;    // FOUT*FIN halves
static const size_t OFF_PARTIAL   = OFF_W_F16 + (size_t)FOUT * FIN * 2; // KSPLIT*256*256 f32

// ---------------------------------------------------------------------------
__device__ __forceinline__ float wave_max32(float v) {
  #pragma unroll
  for (int off = 16; off > 0; off >>= 1)
    v = fmaxf(v, __shfl_xor(v, off, 32));
  return v;
}

__device__ __forceinline__ float block_max256(float v) {
  __shared__ float smax[8];
  v = wave_max32(v);
  const int lane = threadIdx.x & 31, wid = threadIdx.x >> 5;
  if (lane == 0) smax[wid] = v;
  __syncthreads();
  v = (threadIdx.x < 8) ? smax[threadIdx.x] : 0.0f;
  if (wid == 0) v = wave_max32(v);
  return v;  // valid in thread 0
}

__device__ __forceinline__ void atomic_fmax_nonneg(float* addr, float v) {
  // absmax values are >= 0: float bit pattern is monotone as uint32
  atomicMax((unsigned int*)addr, __float_as_uint(v));
}

// Shared conv core: relu((int_conv + b_int) * p_sf * w_sf) for one output elem
__device__ __forceinline__ float conv_relu_at(
    long long tid, const float* __restrict__ x, const float* __restrict__ conv_b,
    float p_sf, const float* __restrict__ wsf, const float* __restrict__ wint) {
  const int ow  = (int)(tid & 31);
  const int oh  = (int)((tid >> 5) & 31);
  const int co  = (int)((tid >> 10) & 63);
  const int n   = (int)(tid >> 16);
  const float inv_p = 1.0f / p_sf;
  float acc = 0.0f;
  #pragma unroll
  for (int ci = 0; ci < CIN; ++ci) {
    const float* xin  = x + ((size_t)n * CIN + ci) * HIN * WIN;
    const float* wrow = wint + (co * CIN + ci) * 9;
    #pragma unroll
    for (int kh = 0; kh < 3; ++kh) {
      const int ih = oh * 2 - 1 + kh;
      if (ih < 0 || ih >= HIN) continue;
      #pragma unroll
      for (int kw = 0; kw < 3; ++kw) {
        const int iw = ow * 2 - 1 + kw;
        if (iw < 0 || iw >= WIN) continue;
        float xi = rintf(xin[ih * WIN + iw] * inv_p);
        xi = fminf(fmaxf(xi, -32767.0f), 32767.0f);
        acc += xi * wrow[kh * 3 + kw];
      }
    }
  }
  const float bias_sf = p_sf * wsf[co];
  const float b_int   = rintf(conv_b[co] / bias_sf);
  return fmaxf((acc + b_int) * bias_sf, 0.0f);
}

// ---------------------------------------------------------------------------
__global__ void k_init(float* scal) {
  if (threadIdx.x < 2) scal[threadIdx.x] = 0.0f;
}

// absmax over x
__global__ void k_absmax(const float* __restrict__ x, long long n, float* dst) {
  long long i = (long long)blockIdx.x * blockDim.x + threadIdx.x;
  const long long stride = (long long)gridDim.x * blockDim.x;
  float m = 0.0f;
  for (; i < n; i += stride) m = fmaxf(m, fabsf(x[i]));
  m = block_max256(m);
  if (threadIdx.x == 0) atomic_fmax_nonneg(dst, m);
}

// Per-output-channel conv weight quantization: 64 blocks x 32 threads
__global__ void k_quant_conv_w(const float* __restrict__ w,
                               float* __restrict__ wsf,
                               float* __restrict__ wint) {
  const int o = blockIdx.x, j = threadIdx.x;       // 27 taps per channel
  float v = (j < 27) ? fabsf(w[o * 27 + j]) : 0.0f;
  float am = wave_max32(v);
  const float sf = fmaxf(am, EPSQ) / 127.0f;
  if (j == 0) wsf[o] = sf;
  if (j < 27) {
    float q = rintf(w[o * 27 + j] / sf);
    wint[o * 27 + j] = fminf(fmaxf(q, -127.0f), 127.0f);
  }
}

// Per-row dense weight quantization -> fp16 ints. One block (256 thr) per row.
__global__ void k_quant_dense_w(const float* __restrict__ w,
                                float* __restrict__ wsf,
                                _Float16* __restrict__ wint) {
  const int o = blockIdx.x;
  const float* row = w + (size_t)o * FIN;
  float m = 0.0f;
  for (int i = threadIdx.x; i < FIN; i += blockDim.x) m = fmaxf(m, fabsf(row[i]));
  m = block_max256(m);
  __shared__ float s_sf;
  if (threadIdx.x == 0) {
    const float sf = fmaxf(m, EPSQ) / 127.0f;
    wsf[o] = sf;
    s_sf = sf;
  }
  __syncthreads();
  const float inv = 1.0f / s_sf;
  _Float16* orow = wint + (size_t)o * FIN;
  for (int i = threadIdx.x; i < FIN; i += blockDim.x) {
    float q = fminf(fmaxf(rintf(row[i] * inv), -127.0f), 127.0f);
    orow[i] = (_Float16)q;
  }
}

// Pass 1: conv -> relu, track global absmax only (no 67MB f32 store)
__global__ void k_conv_absmax(const float* __restrict__ x,
                              const float* __restrict__ conv_b,
                              const float* __restrict__ scal,
                              const float* __restrict__ wsf,
                              const float* __restrict__ wint,
                              float* __restrict__ absmax_h) {
  const long long tid = (long long)blockIdx.x * blockDim.x + threadIdx.x;
  const float p_sf = fmaxf(scal[0], EPSQ) / 32767.0f;
  float v = (tid < HN) ? conv_relu_at(tid, x, conv_b, p_sf, wsf, wint) : 0.0f;
  float bm = block_max256(v);
  if (threadIdx.x == 0) atomic_fmax_nonneg(absmax_h, bm);
}

// Pass 2: recompute conv (x hits L2), quantize at 11 bits -> fp16 ints
__global__ void k_conv_quant(const float* __restrict__ x,
                             const float* __restrict__ conv_b,
                             const float* __restrict__ scal,
                             const float* __restrict__ wsf,
                             const float* __restrict__ wint,
                             _Float16* __restrict__ hq) {
  const long long tid = (long long)blockIdx.x * blockDim.x + threadIdx.x;
  if (tid >= HN) return;
  const float p_sf   = fmaxf(scal[0], EPSQ) / 32767.0f;
  const float scale2 = fmaxf(scal[1], EPSQ) / 1023.0f;
  const float v = conv_relu_at(tid, x, conv_b, p_sf, wsf, wint);
  float q = fminf(fmaxf(rintf(v / scale2), -1023.0f), 1023.0f);
  hq[tid] = (_Float16)q;
}

// Split-K WMMA GEMM partials. Each wave: one 16(M)x32(N) tile, one K chunk.
// A fragment is reused by two WMMAs (halves A re-reads vs 16x16 tiling).
__global__ void k_gemm_splitk(const _Float16* __restrict__ A,   // [256][65536]
                              const _Float16* __restrict__ W,   // [256][65536]
                              float* __restrict__ partial) {    // [KSPLIT][256][256]
  const int lane = threadIdx.x & 31;
  const int wave = threadIdx.x >> 5;
  const int gw   = blockIdx.x * 8 + wave;          // 0..1023
  const int split = gw >> 7;                       // 0..7
  const int t     = gw & 127;                      // 0..127
  const int m0 = (t >> 3) * 16;                    // 16 M tiles
  const int n0 = (t & 7) * 32;                     // 8 N tiles (32 wide)

  // A 16x32 operand, contiguous k order used here (same as B): lanes 0-15 rows M=0..15 K=0..15,
  // lanes 16-31 same rows, K=16..31.
  const int row_m = m0 + (lane & 15);
  const int a_sub = (lane >> 4) * 16;              // 0 or 16
  // B 32x16 operand, contiguous k order used here (same as A): lanes 0-15 cols N=0..15 K=0..15,
  // lanes 16-31 same cols, K=16..31.
  const int b_sub = (lane >> 4) * 16;              // 0 or 16

  const _Float16* __restrict__ arow  = A + (size_t)row_m * FIN;
  const _Float16* __restrict__ wrow0 = W + (size_t)(n0 + (lane & 15)) * FIN;
  const _Float16* __restrict__ wrow1 = wrow0 + (size_t)16 * FIN;

  const int kb = split * KCHUNK, ke = kb + KCHUNK;
  v8f c0 = {}, c1 = {};
  for (int k0 = kb; k0 < ke; k0 += 32) {
    __builtin_prefetch(arow + k0 + 512, 0, 3);     // global_prefetch_b8
    __builtin_prefetch(wrow0 + k0 + 512, 0, 3);
    __builtin_prefetch(wrow1 + k0 + 512, 0, 3);

    v16h a = *(const v16h*)(arow + k0 + a_sub);
    v16h b0 = *(const v16h*)(wrow0 + k0 + b_sub);
    v16h b1 = *(const v16h*)(wrow1 + k0 + b_sub);
    c0 = __builtin_amdgcn_wmma_f32_16x16x32_f16(false, a, false, b0,
                                                (short)0, c0, false, false);
    c1 = __builtin_amdgcn_wmma_f32_16x16x32_f16(false, a, false, b1,
                                                (short)0, c1, false, false);
  }

  // C f32 layout: VGPR r -> M = m0 + r (+8 for lanes 16-31), N = lane&15 col
  float* base = partial + (size_t)split * BATCH * FOUT;
  const int m_base = m0 + ((lane >> 4) * 8);
  const int coln0 = n0 + (lane & 15);
  #pragma unroll
  for (int r = 0; r < 8; ++r) {
    base[(size_t)(m_base + r) * FOUT + coln0]      = c0[r];
    base[(size_t)(m_base + r) * FOUT + coln0 + 16] = c1[r];
  }
}

// Reduce split-K partials + bias/scale/relu epilogue -> out[256,256]
__global__ void k_gemm_reduce(const float* __restrict__ partial,
                              const float* __restrict__ dense_b,
                              const float* __restrict__ scal,
                              const float* __restrict__ wsf,
                              float* __restrict__ out) {
  const int idx = blockIdx.x * blockDim.x + threadIdx.x;  // 65536 elements
  const int o = idx & (FOUT - 1);
  float acc = 0.0f;
  #pragma unroll
  for (int p = 0; p < KSPLIT; ++p)
    acc += partial[(size_t)p * BATCH * FOUT + idx];
  const float scale2  = fmaxf(scal[1], EPSQ) / 1023.0f;
  const float bias_sf = scale2 * wsf[o];
  const float b_int   = rintf(dense_b[o] / bias_sf);
  out[idx] = fmaxf((acc + b_int) * bias_sf, 0.0f);
}

// ---------------------------------------------------------------------------
extern "C" void kernel_launch(void* const* d_in, const int* in_sizes, int n_in,
                              void* d_out, int out_size, void* d_ws, size_t ws_size,
                              hipStream_t stream) {
  const float* x       = (const float*)d_in[0];
  const float* conv_w  = (const float*)d_in[1];
  const float* conv_b  = (const float*)d_in[2];
  const float* dense_w = (const float*)d_in[3];
  const float* dense_b = (const float*)d_in[4];
  float* out = (float*)d_out;

  char* ws = (char*)d_ws;
  float*    scal      = (float*)(ws + OFF_SCAL);
  float*    conv_wsf  = (float*)(ws + OFF_CONV_WSF);
  float*    conv_wint = (float*)(ws + OFF_CONV_WINT);
  float*    dense_wsf = (float*)(ws + OFF_DENSE_WSF);
  _Float16* h_f16     = (_Float16*)(ws + OFF_H_F16);
  _Float16* w_f16     = (_Float16*)(ws + OFF_W_F16);
  float*    partials  = (float*)(ws + OFF_PARTIAL);

  const long long xn = (long long)BATCH * CIN * HIN * WIN;   // 3,145,728

  k_init<<<dim3(1), dim3(32), 0, stream>>>(scal);
  k_absmax<<<dim3(2048), dim3(256), 0, stream>>>(x, xn, &scal[0]);
  k_quant_conv_w<<<dim3(COUT), dim3(32), 0, stream>>>(conv_w, conv_wsf, conv_wint);
  k_quant_dense_w<<<dim3(FOUT), dim3(256), 0, stream>>>(dense_w, dense_wsf, w_f16);
  k_conv_absmax<<<dim3((unsigned)(HN / 256)), dim3(256), 0, stream>>>(
      x, conv_b, scal, conv_wsf, conv_wint, &scal[1]);
  k_conv_quant<<<dim3((unsigned)(HN / 256)), dim3(256), 0, stream>>>(
      x, conv_b, scal, conv_wsf, conv_wint, h_f16);
  k_gemm_splitk<<<dim3(GEMM_WAVES / 8), dim3(256), 0, stream>>>(
      h_f16, w_f16, partials);
  k_gemm_reduce<<<dim3(BATCH * FOUT / 256), dim3(256), 0, stream>>>(
      partials, dense_b, scal, dense_wsf, out);
}
